// SelfAttention_29867202576795
// MI455X (gfx1250) — matrix operand order repaired and checked by reading, not compiled, not run on hardware
//
#include <hip/hip_runtime.h>
#include <hip/hip_bf16.h>
#include <stdint.h>

typedef __attribute__((ext_vector_type(16))) __bf16 v16bf;
typedef __attribute__((ext_vector_type(8)))  float  v8f;
typedef __attribute__((ext_vector_type(4)))  float  f32x4;

#define HD      128
#define NKV     8
#define NQ      32
#define BATCH   32
#define DMODEL  4096
#define SEQ     4096
#define LASTPOS 4095
#define KVROW   (NKV * HD)   /* 1024 floats per cache position */

static __device__ __forceinline__ float4 ldf4(const float* p) {
    return *reinterpret_cast<const float4*>(p);
}
// Non-temporal b128 load (th:TH_LOAD_NT) — keep streamed KV cache out of L2
// so the ~160MB of weights stay resident in MI455X's 192MB L2 across replays.
static __device__ __forceinline__ float4 ldf4_nt(const float* p) {
    f32x4 t = __builtin_nontemporal_load(reinterpret_cast<const f32x4*>(p));
    float4 r; r.x = t.x; r.y = t.y; r.z = t.z; r.w = t.w;
    return r;
}
static __device__ __forceinline__ void put4(float* f, float4 a) {
    f[0] = a.x; f[1] = a.y; f[2] = a.z; f[3] = a.w;
}
static __device__ __forceinline__ v16bf pack16(const float* f) {
    v16bf v;
#pragma unroll
    for (int i = 0; i < 16; ++i) v[i] = (__bf16)f[i];
    return v;
}

// ---------------------------------------------------------------------------
// C[32 x N] = A[32 x K] * W[N x K]^T + bias ; one 16x16 WMMA tile per wave.
// blockDim = 256 (8 waves). grid.x covers N/128, grid.y covers M/16 (=2).
// ---------------------------------------------------------------------------
__global__ __launch_bounds__(256) void wmma_gemm32(
    const float* __restrict__ A, const float* __restrict__ W,
    const float* __restrict__ bias, float* __restrict__ C, int N, int K)
{
    const int wave = threadIdx.x >> 5;
    const int lane = threadIdx.x & 31;
    const int n0 = blockIdx.x * 128 + wave * 16;
    const int m0 = blockIdx.y * 16;
    if (n0 >= N) return;

    const int mlo  = lane & 15;
    const int half = lane >> 4;
    const float* arow = A + (size_t)(m0 + mlo) * K;
    const float* wrow = W + (size_t)(n0 + mlo) * K;

    v8f acc = {};
    for (int k0 = 0; k0 < K; k0 += 32) {
        const float* pa = arow + k0 + half * 8;
        float fa[16];
        put4(fa +  0, ldf4(pa));
        put4(fa +  4, ldf4(pa + 4));
        put4(fa +  8, ldf4(pa + 16));
        put4(fa + 12, ldf4(pa + 20));
        v16bf av = pack16(fa);

        const float* pb = wrow + k0 + half * 8;
        float fb[16];
        put4(fb +  0, ldf4(pb));
        put4(fb +  4, ldf4(pb + 4));
        put4(fb +  8, ldf4(pb + 16));
        put4(fb + 12, ldf4(pb + 20));
        v16bf bv = pack16(fb);

        acc = __builtin_amdgcn_wmma_f32_16x16x32_bf16(
            false, av, false, bv, (short)0, acc, false, false);
    }

    const int n = n0 + mlo;
    const float bvl = bias[n];
#pragma unroll
    for (int j = 0; j < 8; ++j) {
        const int mm = m0 + j + 8 * half;
        C[(size_t)mm * N + n] = acc[j] + bvl;
    }
}

// ---------------------------------------------------------------------------
// RoPE over fresh q (32x32x128) and k (32x8x128), S=1 so cos/sin index = pair.
// ---------------------------------------------------------------------------
__global__ void rope_kernel(float* __restrict__ xq, float* __restrict__ xk,
                            const float* __restrict__ fc, const float* __restrict__ fs)
{
    const int idx = blockIdx.x * blockDim.x + threadIdx.x;  // pair index
    const int total_q = BATCH * NQ * (HD / 2);
    const int total   = total_q + BATCH * NKV * (HD / 2);
    if (idx >= total) return;
    const int p = idx & 63;
    float* base;
    if (idx < total_q) base = xq + (size_t)(idx >> 6) * HD;
    else               base = xk + (size_t)((idx - total_q) >> 6) * HD;
    const float xr = base[2 * p], xi = base[2 * p + 1];
    const float c = fc[p], s = fs[p];
    base[2 * p]     = xr * c - xi * s;
    base[2 * p + 1] = xr * s + xi * c;
}

// ---------------------------------------------------------------------------
// Flash-decode attention. One block per (b, kv_head); 4 waves x 1024 keys.
// QK^T: WMMA (K-dim = HD contiguous, NT b128 loads).
// V tile: async-copied into LDS (global_load_async_to_lds_b128 / ASYNCcnt)
//         while scores+softmax run, then consumed via ds_load_b128.
// ---------------------------------------------------------------------------
__global__ __launch_bounds__(128) void attn_kernel(
    const float* __restrict__ xq, const float* __restrict__ xk,
    const float* __restrict__ xv, const float* __restrict__ cache_k,
    const float* __restrict__ cache_v, float* __restrict__ xo)
{
    const int b    = blockIdx.x >> 3;
    const int h    = blockIdx.x & 7;
    const int wave = threadIdx.x >> 5;
    const int lane = threadIdx.x & 31;
    const int mlo  = lane & 15;
    const int half = lane >> 4;

    __shared__ __align__(16) float vbuf[4][16][HD];     // 32KB: async V tiles
    __shared__ __align__(16) float p_sm[4][4][16];
    __shared__ __align__(16) float comb_o[4][4][HD];
    __shared__ float comb_m[4][4];
    __shared__ float comb_l[4][4];

    // Q fragments (rows = 4 q-heads of this kv-head, padded to 16 with zeros)
    v16bf aq[4];
    {
        const float* qrow = xq + (size_t)b * DMODEL + (size_t)(h * 4 + mlo) * HD;
#pragma unroll
        for (int s = 0; s < 4; ++s) {
            float f[16];
            if (mlo < 4) {
                const float* p0 = qrow + s * 32 + half * 8;
                put4(f +  0, ldf4(p0));
                put4(f +  4, ldf4(p0 + 4));
                put4(f +  8, ldf4(p0 + 16));
                put4(f + 12, ldf4(p0 + 20));
            } else {
#pragma unroll
                for (int e = 0; e < 16; ++e) f[e] = 0.f;
            }
            aq[s] = pack16(f);
        }
    }

    const float sc = 0.08838834764831845f;  // 1/sqrt(128)
    float  m_st[4] = {-__builtin_inff(), -__builtin_inff(),
                      -__builtin_inff(), -__builtin_inff()};
    float  l_st[4] = {0.f, 0.f, 0.f, 0.f};
    float4 o_st[4];
#pragma unroll
    for (int r = 0; r < 4; ++r) { o_st[r].x = o_st[r].y = o_st[r].z = o_st[r].w = 0.f; }

    const float* knew = xk + (size_t)(b * NKV + h) * HD;
    const float* vnew = xv + (size_t)(b * NKV + h) * HD;
    const size_t cbase = (size_t)b * SEQ * KVROW + (size_t)h * HD;

    // per-lane LDS byte address of this wave's V staging buffer
    const uint32_t vbuf_lane = (uint32_t)(uintptr_t)&vbuf[wave][0][0]
                             + (uint32_t)(lane * 16);

    const int jbeg = wave * 1024, jend = jbeg + 1024;
    for (int j0 = jbeg; j0 < jend; j0 += 16) {
        // ---- kick off async copy of the V tile (16 rows x 512B) to LDS ----
        // previous tile's ds reads must retire before the async engine
        // overwrites the single-buffered tile (async is unordered vs DS).
        asm volatile("s_wait_dscnt 0x0" ::: "memory");
#pragma unroll
        for (int k = 0; k < 16; ++k) {
            const int j = j0 + k;
            const float* vrow = (j == LASTPOS) ? vnew
                              : (cache_v + cbase + (size_t)j * KVROW);
            const uint64_t ga = (uint64_t)(uintptr_t)vrow + (uint64_t)(lane * 16);
            const uint32_t la = vbuf_lane + (uint32_t)(k * 512);
            asm volatile("global_load_async_to_lds_b128 %0, %1, off"
                         :: "v"(la), "v"(ga) : "memory");
        }

        // ---- scores for 16 keys via 4 WMMAs (overlaps with async V copy) ---
        const int jkey = j0 + mlo;
        const float* krow = (jkey == LASTPOS) ? knew
                          : (cache_k + cbase + (size_t)jkey * KVROW);
        if (jkey + 16 < SEQ)
            __builtin_prefetch(cache_k + cbase + (size_t)(jkey + 16) * KVROW, 0, 1);

        v8f cfrag = {};
#pragma unroll
        for (int s = 0; s < 4; ++s) {
            const float* pb = krow + s * 32 + half * 8;
            float fb[16];
            put4(fb +  0, ldf4_nt(pb));
            put4(fb +  4, ldf4_nt(pb + 4));
            put4(fb +  8, ldf4_nt(pb + 16));
            put4(fb + 12, ldf4_nt(pb + 20));
            v16bf bv = pack16(fb);
            cfrag = __builtin_amdgcn_wmma_f32_16x16x32_bf16(
                false, aq[s], false, bv, (short)0, cfrag, false, false);
        }

        // ---- online softmax, rows 0..3 live in vgprs 0..3 of half==0 lanes
#pragma unroll
        for (int r = 0; r < 4; ++r) {
            float val = (half == 0) ? cfrag[r] * sc : -__builtin_inff();
            float mx = val;
#pragma unroll
            for (int ofs = 16; ofs >= 1; ofs >>= 1)
                mx = fmaxf(mx, __shfl_xor(mx, ofs, 32));
            const float mnew  = fmaxf(m_st[r], mx);
            const float alpha = __expf(m_st[r] - mnew);
            const float p     = (half == 0) ? __expf(val - mnew) : 0.f;
            float ssum = p;
#pragma unroll
            for (int ofs = 16; ofs >= 1; ofs >>= 1)
                ssum += __shfl_xor(ssum, ofs, 32);
            l_st[r] = l_st[r] * alpha + ssum;
            m_st[r] = mnew;
            o_st[r].x *= alpha; o_st[r].y *= alpha;
            o_st[r].z *= alpha; o_st[r].w *= alpha;
            if (half == 0) p_sm[wave][r][mlo] = p;
        }
        __syncthreads();

        // ---- P * V from the async-staged LDS tile ----
        asm volatile("s_wait_asynccnt 0x0" ::: "memory");
#pragma unroll 4
        for (int k = 0; k < 16; ++k) {
            const float4 v4 = *reinterpret_cast<const float4*>(&vbuf[wave][k][4 * lane]);
#pragma unroll
            for (int r = 0; r < 4; ++r) {
                const float p = p_sm[wave][r][k];
                o_st[r].x += p * v4.x; o_st[r].y += p * v4.y;
                o_st[r].z += p * v4.z; o_st[r].w += p * v4.w;
            }
        }
        __syncthreads();
    }

    // ---- merge the 4 waves' partial (m, l, O) ----
#pragma unroll
    for (int r = 0; r < 4; ++r)
        *reinterpret_cast<float4*>(&comb_o[wave][r][4 * lane]) = o_st[r];
    if (lane == 0) {
#pragma unroll
        for (int r = 0; r < 4; ++r) { comb_m[wave][r] = m_st[r]; comb_l[wave][r] = l_st[r]; }
    }
    __syncthreads();

    const int r  = threadIdx.x >> 5;   // 0..3 (q-head within group)
    const int d4 = threadIdx.x & 31;   // dim/4
    float M = comb_m[0][r];
#pragma unroll
    for (int w = 1; w < 4; ++w) M = fmaxf(M, comb_m[w][r]);
    float L = 0.f;
    float4 acc; acc.x = acc.y = acc.z = acc.w = 0.f;
#pragma unroll
    for (int w = 0; w < 4; ++w) {
        const float e = __expf(comb_m[w][r] - M);
        L += comb_l[w][r] * e;
        const float4 ov = *reinterpret_cast<const float4*>(&comb_o[w][r][4 * d4]);
        acc.x += ov.x * e; acc.y += ov.y * e; acc.z += ov.z * e; acc.w += ov.w * e;
    }
    const float inv = 1.f / L;
    float* dst = xo + (size_t)b * DMODEL + (size_t)(h * 4 + r) * HD + 4 * d4;
    dst[0] = acc.x * inv; dst[1] = acc.y * inv;
    dst[2] = acc.z * inv; dst[3] = acc.w * inv;
}

// ---------------------------------------------------------------------------
extern "C" void kernel_launch(void* const* d_in, const int* in_sizes, int n_in,
                              void* d_out, int out_size, void* d_ws, size_t ws_size,
                              hipStream_t stream) {
    (void)in_sizes; (void)n_in; (void)out_size; (void)ws_size;
    const float* x       = (const float*)d_in[0];
    // d_in[1] = start_pos (compile-time constant 4095 in the reference)
    const float* fc      = (const float*)d_in[2];
    const float* fs      = (const float*)d_in[3];
    const float* wq_w    = (const float*)d_in[4];
    const float* wq_b    = (const float*)d_in[5];
    const float* wk_w    = (const float*)d_in[6];
    const float* wk_b    = (const float*)d_in[7];
    const float* wv_w    = (const float*)d_in[8];
    const float* wv_b    = (const float*)d_in[9];
    const float* wo_w    = (const float*)d_in[10];
    const float* wo_b    = (const float*)d_in[11];
    const float* cache_k = (const float*)d_in[12];
    const float* cache_v = (const float*)d_in[13];
    float* out = (float*)d_out;

    float* ws = (float*)d_ws;
    float* xq = ws;                       // 32*4096
    float* xk = xq + BATCH * DMODEL;      // 32*1024
    float* xv = xk + BATCH * NKV * HD;    // 32*1024
    float* xo = xv + BATCH * NKV * HD;    // 32*4096

    const dim3 blk(256);
    wmma_gemm32<<<dim3(32, 2), blk, 0, stream>>>(x, wq_w, wq_b, xq, DMODEL, DMODEL);
    wmma_gemm32<<<dim3(8, 2),  blk, 0, stream>>>(x, wk_w, wk_b, xk, NKV * HD, DMODEL);
    wmma_gemm32<<<dim3(8, 2),  blk, 0, stream>>>(x, wv_w, wv_b, xv, NKV * HD, DMODEL);
    rope_kernel<<<dim3(320), blk, 0, stream>>>(xq, xk, fc, fs);
    attn_kernel<<<dim3(256), dim3(128), 0, stream>>>(xq, xk, xv, cache_k, cache_v, xo);
    wmma_gemm32<<<dim3(32, 2), blk, 0, stream>>>(xo, wo_w, wo_b, out, DMODEL, DMODEL);
}
